// SPINN_63548336112109
// MI455X (gfx1250) — matrix operand order repaired and checked by reading, not compiled, not run on hardware
//
#include <hip/hip_runtime.h>

// ---------------- problem constants ----------------
#define Bn   128
#define Ln   48
#define Dn   256
#define TRn  256
#define Tn   93          // 2*L - 3
#define SMn  50          // stack S_max = L + 2
#define NWG  16
#define NTHR 256         // 8 waves per workgroup -> 128 waves total

typedef __bf16 bf16_t;
typedef __attribute__((ext_vector_type(16))) __bf16 v16bf;
typedef __attribute__((ext_vector_type(8)))  __bf16 v8bf;
typedef __attribute__((ext_vector_type(8)))  float  v8f;

// ---------------- workspace layout (bytes, all 16B aligned) ----------------
constexpr size_t O_WIH = 0;                                   // 1024x768 bf16
constexpr size_t O_WHH = O_WIH + (size_t)1024*768*2;          // 1024x256 bf16
constexpr size_t O_WL  = O_WHH + (size_t)1024*256*2;          // 1280x256 bf16
constexpr size_t O_WR  = O_WL  + (size_t)1280*256*2;
constexpr size_t O_WT  = O_WR  + (size_t)1280*256*2;
constexpr size_t O_BG  = O_WT  + (size_t)1280*256*2;          // 1024 f32 (b_ih+b_hh)
constexpr size_t O_BL  = O_BG  + (size_t)1024*4;              // 1280 f32
constexpr size_t O_X   = O_BL  + (size_t)1280*4;              // 128x768 bf16
constexpr size_t O_TH0 = O_X   + (size_t)Bn*768*2;            // 128x256 bf16 (ping)
constexpr size_t O_TH1 = O_TH0 + (size_t)Bn*TRn*2;            // 128x256 bf16 (pong)
constexpr size_t O_TC  = O_TH1 + (size_t)Bn*TRn*2;            // 128x256 f32
constexpr size_t O_STK = O_TC  + (size_t)Bn*TRn*4;            // 128x50x512 f32
constexpr size_t O_BP  = O_STK + (size_t)Bn*SMn*512*4;        // T*B i32 each:
constexpr size_t O_S1  = O_BP  + (size_t)Tn*Bn*4;
constexpr size_t O_S2  = O_S1  + (size_t)Tn*Bn*4;
constexpr size_t O_WP  = O_S2  + (size_t)Tn*Bn*4;
constexpr size_t O_SH  = O_WP  + (size_t)Tn*Bn*4;
constexpr size_t O_FP  = O_SH  + (size_t)Tn*Bn*4;             // final ptr per b
constexpr size_t O_BAR = O_FP  + (size_t)Bn*4 + 256;          // barrier cnt/gen

// ---------------- helpers ----------------
__device__ __forceinline__ float sigf(float x) { return 1.0f / (1.0f + __expf(-x)); }

// A fragment: 16(M) x 32(K) bf16 from row-major src (ld elements per row).
// Lanes 0-15: M=lane, K in {0..7, 16..23}; lanes 16-31: M=lane-16, K in {8..15, 24..31}.
__device__ __forceinline__ v16bf load_frag_a(const bf16_t* __restrict__ src, int ld,
                                             int m0, int k0) {
  const int lane = threadIdx.x & 31;
  const int m  = m0 + (lane & 15);
  const int kb = (lane & 16) ? 8 : 0;
  const bf16_t* p = src + (size_t)m * ld + k0 + kb;
  union { v16bf v; v8bf h[2]; } u;
  u.h[0] = *(const v8bf*)(p);        // K = kb .. kb+7
  u.h[1] = *(const v8bf*)(p + 16);   // K = 16+kb .. 16+kb+7
  return u.v;
}

// B fragment: 32(K) x 16(N) bf16 where the source is W[n][k] row-major (ld = K stride).
// Lanes 0-15: N=lane, K in {0..7, 16..23}; lanes 16-31: N=lane-16, K in {8..15, 24..31}.
__device__ __forceinline__ v16bf load_frag_b(const bf16_t* __restrict__ w, int ld,
                                             int n0, int k0) {
  const int lane = threadIdx.x & 31;
  const int n  = n0 + (lane & 15);
  const int kb = (lane & 16) ? 8 : 0;
  const bf16_t* p = w + (size_t)n * ld + k0 + kb;
  union { v16bf v; v8bf h[2]; } u;
  u.h[0] = *(const v8bf*)(p);
  u.h[1] = *(const v8bf*)(p + 16);
  return u.v;
}

__device__ __forceinline__ v8f wmma_bf16(v16bf a, v16bf b, v8f c) {
  return __builtin_amdgcn_wmma_f32_16x16x32_bf16(false, a, false, b, (short)0, c,
                                                 false, false);
}

// Grid-wide barrier (sense via generation counter). All blocks must call it.
__device__ __forceinline__ void gbar(unsigned* cnt, unsigned* gen) {
  __threadfence();            // make this thread's global stores visible device-wide
  __syncthreads();
  if (threadIdx.x == 0) {
    volatile unsigned* vgen = (volatile unsigned*)gen;
    unsigned g = *vgen;
    unsigned a = atomicAdd(cnt, 1u);
    if (a == NWG - 1) {
      atomicExch(cnt, 0u);
      __threadfence();
      atomicAdd(gen, 1u);
    } else {
      while (*vgen == g) { __builtin_amdgcn_s_sleep(2); }
    }
    __threadfence();
  }
  __syncthreads();
}

// ---------------- prep kernels ----------------
// Precompute per-(t,b) stack indices from transitions; zero the barrier.
__global__ void spinn_idx(const int* __restrict__ trans, int* bp, int* s1, int* s2,
                          int* wp, int* sh, int* fp, unsigned* bar) {
  int b = threadIdx.x;
  if (b == 0) { bar[0] = 0u; bar[64] = 0u; }
  if (b < Bn) {
    int ptr = 2, bptr = Ln - 1;
    for (int t = 0; t < Tn; ++t) {
      int is = (trans[t * Bn + b] == 3);   // SHIFT
      bp[t * Bn + b] = bptr;
      s1[t * Bn + b] = ptr - 1;
      s2[t * Bn + b] = ptr - 2;
      wp[t * Bn + b] = is ? ptr : ptr - 2;
      sh[t * Bn + b] = is;
      ptr += is ? 1 : -1;
      bptr -= is;
    }
    fp[b] = ptr;
  }
}

// Convert weights f32 -> bf16 (layout preserved: W[out][in] == B-matrix [n][k]).
__global__ void spinn_prep_w(const float* __restrict__ wl, const float* __restrict__ wr,
                             const float* __restrict__ wt, const float* __restrict__ wih,
                             const float* __restrict__ whh, const float* __restrict__ bih,
                             const float* __restrict__ bhh, const float* __restrict__ blf,
                             bf16_t* Wih, bf16_t* Whh, bf16_t* Wl, bf16_t* Wr, bf16_t* Wt,
                             float* bg, float* bl) {
  int gt = blockIdx.x * blockDim.x + threadIdx.x;
  int gs = gridDim.x * blockDim.x;
  for (int i = gt; i < 1024 * 768; i += gs) Wih[i] = (bf16_t)wih[i];
  for (int i = gt; i < 1024 * 256; i += gs) Whh[i] = (bf16_t)whh[i];
  for (int i = gt; i < 1280 * 256; i += gs) {
    Wl[i] = (bf16_t)wl[i]; Wr[i] = (bf16_t)wr[i]; Wt[i] = (bf16_t)wt[i];
  }
  for (int i = gt; i < 1024; i += gs) bg[i] = bih[i] + bhh[i];
  for (int i = gt; i < 1280; i += gs) bl[i] = blf[i];
}

// Zero recurrent state, init stack rows 0/1 with buffers[:,0].
__global__ void spinn_init(const float* __restrict__ buffers, float* tc,
                           bf16_t* th0, bf16_t* th1, float* stack) {
  int gt = blockIdx.x * blockDim.x + threadIdx.x;
  int gs = gridDim.x * blockDim.x;
  for (int i = gt; i < Bn * TRn; i += gs) {
    tc[i] = 0.0f; th0[i] = (bf16_t)0.0f; th1[i] = (bf16_t)0.0f;
  }
  for (int i = gt; i < Bn * 512; i += gs) {
    int b = i >> 9, k = i & 511;
    float v = buffers[(size_t)b * Ln * 512 + k];          // buffers[b][0][k]
    stack[((size_t)b * SMn + 0) * 512 + k] = v;
    stack[((size_t)b * SMn + 1) * 512 + k] = v;
  }
}

// ---------------- persistent main kernel ----------------
__global__ void __launch_bounds__(NTHR, 1)
spinn_main(const float* __restrict__ buffers,
           const bf16_t* __restrict__ Wih, const bf16_t* __restrict__ Whh,
           const bf16_t* __restrict__ Wl,  const bf16_t* __restrict__ Wr,
           const bf16_t* __restrict__ Wt,
           const float* __restrict__ bg,   const float* __restrict__ bl,
           bf16_t* __restrict__ xbuf, bf16_t* th0, bf16_t* th1,
           float* __restrict__ tc, float* __restrict__ stack,
           const int* __restrict__ bp, const int* __restrict__ s1i,
           const int* __restrict__ s2i, const int* __restrict__ wpi,
           const int* __restrict__ shi, const int* __restrict__ fp,
           unsigned* bar, float* __restrict__ out) {
  const int tid   = threadIdx.x;
  const int gtid  = blockIdx.x * NTHR + tid;
  const int gsize = NWG * NTHR;
  const int w     = gtid >> 5;                  // global wave id, 0..127
  const int lane  = tid & 31;

  unsigned* cnt = bar;
  unsigned* gen = bar + 64;

  const int m0 = (w >> 4) << 4;                 // 8 m-tiles (batch rows)
  const int j0 = (w & 15) << 4;                 // 16 column tiles within TR/D
  const int n  = j0 + (lane & 15);              // this lane's output column
  const int rbase = m0 + ((lane >> 4) << 3);    // C/D frag row base

  bf16_t* thbuf[2] = { th0, th1 };
  const v8f vzero = {0.f, 0.f, 0.f, 0.f, 0.f, 0.f, 0.f, 0.f};

  for (int t = 0; t < Tn; ++t) {
    const int* bp_t = bp  + t * Bn;
    const int* s1_t = s1i + t * Bn;
    const int* s2_t = s2i + t * Bn;

    // ---- Phase A: x = bf16(concat(buf_top_h, s1_h, s2_h)) ----
    for (int idx = gtid; idx < Bn * Dn; idx += gsize) {
      int b = idx >> 8, k = idx & 255;
      float bt = buffers[((size_t)b * Ln + bp_t[b]) * 512 + k];
      float v1 = stack[((size_t)b * SMn + s1_t[b]) * 512 + k];
      float v2 = stack[((size_t)b * SMn + s2_t[b]) * 512 + k];
      bf16_t* xr = xbuf + (size_t)b * 768;
      xr[k]       = (bf16_t)bt;
      xr[256 + k] = (bf16_t)v1;   // rh = s1 h
      xr[512 + k] = (bf16_t)v2;   // lh = s2 h
    }
    gbar(cnt, gen);

    // ---- Phase B: gates GEMM (x@Wih^T + th@Whh^T) + tracker LSTM pointwise ----
    const bf16_t* thin  = thbuf[t & 1];
    bf16_t*       thout = thbuf[(t + 1) & 1];
    v8f acc[4] = {vzero, vzero, vzero, vzero};
    for (int k0 = 0; k0 < 768; k0 += 32) {
      v16bf a = load_frag_a(xbuf, 768, m0, k0);
#pragma unroll
      for (int g = 0; g < 4; ++g)
        acc[g] = wmma_bf16(a, load_frag_b(Wih, 768, g * 256 + j0, k0), acc[g]);
    }
    for (int k0 = 0; k0 < 256; k0 += 32) {
      v16bf a = load_frag_a(thin, 256, m0, k0);
#pragma unroll
      for (int g = 0; g < 4; ++g)
        acc[g] = wmma_bf16(a, load_frag_b(Whh, 256, g * 256 + j0, k0), acc[g]);
    }
    {
      const float bi = bg[n], bf_ = bg[256 + n], bgg = bg[512 + n], bo = bg[768 + n];
#pragma unroll
      for (int r = 0; r < 8; ++r) {
        int b = rbase + r;
        float gi = acc[0][r] + bi;
        float gf = acc[1][r] + bf_;
        float gg = acc[2][r] + bgg;
        float go = acc[3][r] + bo;
        float tco = tc[b * TRn + n];
        float tcn = sigf(gf) * tco + sigf(gi) * tanhf(gg);
        float thn = sigf(go) * tanhf(tcn);
        tc[b * TRn + n]    = tcn;            // exclusive (b,n) ownership
        thout[b * TRn + n] = (bf16_t)thn;    // ping-pong: no reader race
      }
    }
    gbar(cnt, gen);

    // ---- Phase D: reduce GEMM (lh@Wl^T + rh@Wr^T + thn@Wt^T) + TreeLSTM + stack op ----
    v8f acc5[5] = {vzero, vzero, vzero, vzero, vzero};
    for (int k0 = 0; k0 < 256; k0 += 32) {
      v16bf al = load_frag_a(xbuf, 768, m0, 512 + k0);   // lh (s2 h)
      v16bf ar = load_frag_a(xbuf, 768, m0, 256 + k0);   // rh (s1 h)
      v16bf at = load_frag_a(thout, 256, m0, k0);        // th_n
#pragma unroll
      for (int g = 0; g < 5; ++g) {
        acc5[g] = wmma_bf16(al, load_frag_b(Wl, 256, g * 256 + j0, k0), acc5[g]);
        acc5[g] = wmma_bf16(ar, load_frag_b(Wr, 256, g * 256 + j0, k0), acc5[g]);
        acc5[g] = wmma_bf16(at, load_frag_b(Wt, 256, g * 256 + j0, k0), acc5[g]);
      }
    }
    {
      const int* wp_t = wpi + t * Bn;
      const int* sh_t = shi + t * Bn;
      const float ba = bl[n], bi2 = bl[256 + n], bf1 = bl[512 + n],
                  bf2 = bl[768 + n], bo2 = bl[1024 + n];
#pragma unroll
      for (int r = 0; r < 8; ++r) {
        int b = rbase + r;
        float av = tanhf(acc5[0][r] + ba);
        float iv = sigf(acc5[1][r] + bi2);
        float f1 = sigf(acc5[2][r] + bf1);
        float f2 = sigf(acc5[3][r] + bf2);
        float ov = sigf(acc5[4][r] + bo2);
        float lc = stack[((size_t)b * SMn + s2_t[b]) * 512 + 256 + n];
        float rc = stack[((size_t)b * SMn + s1_t[b]) * 512 + 256 + n];
        float c  = av * iv + f1 * lc + f2 * rc;
        float h  = ov * tanhf(c);
        float* dst = stack + ((size_t)b * SMn + wp_t[b]) * 512;
        if (sh_t[b]) {
          const float* btr = buffers + ((size_t)b * Ln + bp_t[b]) * 512;
          dst[n]       = btr[n];
          dst[256 + n] = btr[256 + n];
        } else {
          dst[n]       = h;
          dst[256 + n] = c;
        }
      }
    }
    gbar(cnt, gen);
  }

  // ---- Output: stack[b][final_ptr-1] ----
  for (int idx = gtid; idx < Bn * 512; idx += gsize) {
    int b = idx >> 9, k = idx & 511;
    out[idx] = stack[((size_t)b * SMn + (fp[b] - 1)) * 512 + k];
  }
}

// ---------------- host launcher ----------------
extern "C" void kernel_launch(void* const* d_in, const int* in_sizes, int n_in,
                              void* d_out, int out_size, void* d_ws, size_t ws_size,
                              hipStream_t stream) {
  (void)in_sizes; (void)n_in; (void)out_size; (void)ws_size;
  const float* buffers = (const float*)d_in[0];
  const int*   trans   = (const int*)d_in[1];
  const float* W_left  = (const float*)d_in[2];
  const float* b_left  = (const float*)d_in[3];
  const float* W_right = (const float*)d_in[4];
  const float* W_track = (const float*)d_in[5];
  const float* W_ih    = (const float*)d_in[6];
  const float* W_hh    = (const float*)d_in[7];
  const float* b_ih    = (const float*)d_in[8];
  const float* b_hh    = (const float*)d_in[9];

  char* ws = (char*)d_ws;
  bf16_t* Wih = (bf16_t*)(ws + O_WIH);
  bf16_t* Whh = (bf16_t*)(ws + O_WHH);
  bf16_t* Wl  = (bf16_t*)(ws + O_WL);
  bf16_t* Wr  = (bf16_t*)(ws + O_WR);
  bf16_t* Wt  = (bf16_t*)(ws + O_WT);
  float*  bg  = (float*)(ws + O_BG);
  float*  bl  = (float*)(ws + O_BL);
  bf16_t* xb  = (bf16_t*)(ws + O_X);
  bf16_t* th0 = (bf16_t*)(ws + O_TH0);
  bf16_t* th1 = (bf16_t*)(ws + O_TH1);
  float*  tc  = (float*)(ws + O_TC);
  float*  stk = (float*)(ws + O_STK);
  int* bp = (int*)(ws + O_BP);
  int* s1 = (int*)(ws + O_S1);
  int* s2 = (int*)(ws + O_S2);
  int* wp = (int*)(ws + O_WP);
  int* sh = (int*)(ws + O_SH);
  int* fp = (int*)(ws + O_FP);
  unsigned* bar = (unsigned*)(ws + O_BAR);

  spinn_idx<<<1, 128, 0, stream>>>(trans, bp, s1, s2, wp, sh, fp, bar);
  spinn_prep_w<<<256, 256, 0, stream>>>(W_left, W_right, W_track, W_ih, W_hh,
                                        b_ih, b_hh, b_left,
                                        Wih, Whh, Wl, Wr, Wt, bg, bl);
  spinn_init<<<128, 256, 0, stream>>>(buffers, tc, th0, th1, stk);
  spinn_main<<<NWG, NTHR, 0, stream>>>(buffers, Wih, Whh, Wl, Wr, Wt, bg, bl,
                                       xb, th0, th1, tc, stk,
                                       bp, s1, s2, wp, sh, fp, bar, (float*)d_out);
}
